// kan_embedding_37898791420535
// MI455X (gfx1250) — matrix operand order repaired and checked by reading, not compiled, not run on hardware
//
#include <hip/hip_runtime.h>

// ---------------------------------------------------------------------------
// KAN autoencoder for MI455X (gfx1250, wave32).
// Big layers (E->D, D->E) are cast to a single f16 WMMA GEMM with K = 9*in:
// each input feature contributes 9 channels [silu(x), B-spline basis 0..7],
// and the weight matrix is the matching concat [base_w, spline_w*scaler].
// Accumulation in f32 via v_wmma_f32_16x16x32_f16.
// GEMM uses 4x4 register blocking: 64x64 output per wave, 16 WMMAs per
// 16 b128 loads per K-step (1.0 load/WMMA).
// ---------------------------------------------------------------------------

typedef __attribute__((ext_vector_type(16))) _Float16 v16h;
typedef __attribute__((ext_vector_type(8)))  _Float16 v8h;
typedef __attribute__((ext_vector_type(8)))  float    v8f;

#define KAN_NCOEF 8
#define KAN_HSTEP 0.4f   // (1 - (-1)) / GRID_SIZE(5)

// Cox-de Boor recursion on the uniform extended knot vector
// t_j = -1 + (j-3)*h, j = 0..11  (GRID_SIZE=5, SPLINE_ORDER=3)
__device__ __forceinline__ void kan_bspline8(float x, float bs[8]) {
  float t[12];
#pragma unroll
  for (int j = 0; j < 12; ++j) t[j] = -1.0f + (float)(j - 3) * KAN_HSTEP;
  float b[11];
#pragma unroll
  for (int j = 0; j < 11; ++j) b[j] = (x >= t[j] && x < t[j + 1]) ? 1.0f : 0.0f;
#pragma unroll
  for (int k = 1; k <= 3; ++k) {
    const float inv = 1.0f / ((float)k * KAN_HSTEP);
#pragma unroll
    for (int j = 0; j + k < 11; ++j) {
      float left  = (x - t[j]) * inv * b[j];
      float right = (t[j + k + 1] - x) * inv * b[j + 1];
      b[j] = left + right;
    }
  }
#pragma unroll
  for (int j = 0; j < 8; ++j) bs[j] = b[j];
}

__device__ __forceinline__ float kan_silu(float x) {
  return x / (1.0f + __expf(-x));
}
__device__ __forceinline__ float kan_leaky(float v) {
  return v >= 0.0f ? v : 0.1f * v;
}

// ---------------------------------------------------------------------------
// Build augmented weight matrix Waug (OUT x 9*IN, f16, row-major):
//   Waug[o][0*IN + i]     = base_w[o][i]
//   Waug[o][(c+1)*IN + i] = spline_w[o][i][c] * scaler[o][i]
// ---------------------------------------------------------------------------
__global__ void kan_build_w_kernel(const float* __restrict__ baseW,
                                   const float* __restrict__ splineW,
                                   const float* __restrict__ scaler,
                                   _Float16* __restrict__ Waug,
                                   int OUT, int IN) {
  int idx = blockIdx.x * blockDim.x + threadIdx.x;
  if (idx >= OUT * IN) return;
  int o = idx / IN, i = idx % IN;
  size_t rb = (size_t)o * (size_t)(9 * IN);
  float sc = scaler[idx];
  Waug[rb + i] = (_Float16)baseW[idx];
#pragma unroll
  for (int c = 0; c < KAN_NCOEF; ++c)
    Waug[rb + (size_t)(c + 1) * IN + i] =
        (_Float16)(splineW[(size_t)idx * KAN_NCOEF + c] * sc);
}

// ---------------------------------------------------------------------------
// Build augmented activation matrix A (N x 9*IN, f16, row-major):
//   A[n][0*IN + i]     = silu(x[n][i])
//   A[n][(c+1)*IN + i] = basis_c(x[n][i])
// ---------------------------------------------------------------------------
__global__ void kan_build_a_kernel(const float* __restrict__ X,
                                   _Float16* __restrict__ A,
                                   int N, int IN) {
  int idx = blockIdx.x * blockDim.x + threadIdx.x;
  if (idx >= N * IN) return;
  int n = idx / IN, i = idx % IN;
  float x = X[idx];
  float bs[8];
  kan_bspline8(x, bs);
  size_t rb = (size_t)n * (size_t)(9 * IN);
  A[rb + i] = (_Float16)kan_silu(x);
#pragma unroll
  for (int c = 0; c < KAN_NCOEF; ++c)
    A[rb + (size_t)(c + 1) * IN + i] = (_Float16)bs[c];
}

// ---------------------------------------------------------------------------
// WMMA GEMM: C[N x M] = A[N x K] * Waug^T   (Waug is M x K row-major)
// Wave tile 64x64 (4x4 accumulators), workgroup tile 128x256 (8 waves, 2x4).
// Fragment layouts used here (wave32; A/B use contiguous k, not the ISA table order):
//   A frag : lane L = row L%16; halves j  at k=16*(L/16)+j  (contiguous, matches B)
//   B frag : lane L = col L%16; halves j  at k=16*(L/16)+j  (contiguous in Waug)
//   D frag : VGPR r, lane L -> row r+8*(L/16), col L%16
// ---------------------------------------------------------------------------
__global__ __launch_bounds__(256)
void kan_wmma_gemm_kernel(const _Float16* __restrict__ A,
                          const _Float16* __restrict__ W,
                          float* __restrict__ C,
                          int N, int M, int K, int applyLeaky) {
  const int lane = threadIdx.x & 31;
  const int wv   = threadIdx.x >> 5;
  const int g    = lane >> 4;             // half-wave group (0/1)
  const int l    = lane & 15;

  // Wave tile placement: 2x4 wave grid inside the block.
  const int nt0 = blockIdx.y * 8  + (wv >> 2) * 4;  // 16-row tile index
  const int mt0 = blockIdx.x * 16 + (wv & 3) * 4;   // 16-col tile index

  const _Float16* pa[4];
  const _Float16* pb[4];
#pragma unroll
  for (int i = 0; i < 4; ++i)
    pa[i] = A + (size_t)((nt0 + i) * 16 + l) * (size_t)K + g * 16;
#pragma unroll
  for (int j = 0; j < 4; ++j)
    pb[j] = W + (size_t)((mt0 + j) * 16 + l) * (size_t)K + g * 16;

  v8f acc[16] = {};

  for (int kt = 0; kt < K; kt += 32) {
    union { v16h v; v8h h[2]; } a[4];
    v16h b[4];
#pragma unroll
    for (int i = 0; i < 4; ++i) {
      a[i].h[0] = *(const v8h*)(pa[i] + kt);
      a[i].h[1] = *(const v8h*)(pa[i] + kt + 8);
    }
#pragma unroll
    for (int j = 0; j < 4; ++j)
      b[j] = *(const v16h*)(pb[j] + kt);

#pragma unroll
    for (int i = 0; i < 4; ++i)
#pragma unroll
      for (int j = 0; j < 4; ++j)
        acc[i * 4 + j] = __builtin_amdgcn_wmma_f32_16x16x32_f16(
            /*neg_a=*/false, a[i].v, /*neg_b=*/false, b[j],
            /*c_mod=*/(short)0, acc[i * 4 + j],
            /*reuse_a=*/false, /*reuse_b=*/false);
  }

#pragma unroll
  for (int i = 0; i < 4; ++i) {
    const int row0 = (nt0 + i) * 16 + g * 8;
#pragma unroll
    for (int j = 0; j < 4; ++j) {
      const int col = (mt0 + j) * 16 + l;
#pragma unroll
      for (int r = 0; r < 8; ++r) {
        float v = acc[i * 4 + j][r];
        if (applyLeaky) v = kan_leaky(v);
        C[(size_t)(row0 + r) * (size_t)M + col] = v;
      }
    }
  }
}

// ---------------------------------------------------------------------------
// Fused middle layers (tiny): for each (b,d) of B*D rows:
//   layer 2 (H -> 1): u = sum_h [ silu(x_emb[b,h,d])*w2b[h]
//                                + sum_c basis_c(x_emb[b,h,d])*w2s[h,c]*w2c[h] ]
//   store x_comb[b*D+d] = u   (this is x_comb.transpose(0,2,1) flat)
//   layer 3 (1 -> H): x_de[b,h,d] = leaky( silu(u)*w3b[h]
//                                + sum_c basis_c(u)*w3s[h,c]*w3c[h] )
// ---------------------------------------------------------------------------
__global__ void kan_mid_kernel(const float* __restrict__ x_emb,
                               const float* __restrict__ w2b,
                               const float* __restrict__ w2s,
                               const float* __restrict__ w2c,
                               const float* __restrict__ w3b,
                               const float* __restrict__ w3s,
                               const float* __restrict__ w3c,
                               float* __restrict__ x_comb,
                               float* __restrict__ x_de,
                               int B, int H, int D) {
  int idx = blockIdx.x * blockDim.x + threadIdx.x;
  if (idx >= B * D) return;
  int b = idx / D, d = idx % D;

  const float* xe = x_emb + (size_t)b * H * D + d;
  float u = 0.0f;
  for (int h = 0; h < H; ++h) {
    float tv = xe[(size_t)h * D];
    float bs[8];
    kan_bspline8(tv, bs);
    float acc = kan_silu(tv) * w2b[h];
    float sc = w2c[h];
#pragma unroll
    for (int c = 0; c < KAN_NCOEF; ++c) acc += bs[c] * w2s[h * KAN_NCOEF + c] * sc;
    u += acc;
  }
  x_comb[idx] = u;

  float su = kan_silu(u);
  float bu[8];
  kan_bspline8(u, bu);
  float* xd = x_de + (size_t)b * H * D + d;
  for (int h = 0; h < H; ++h) {
    float v = su * w3b[h];
    float sc = w3c[h];
#pragma unroll
    for (int c = 0; c < KAN_NCOEF; ++c) v += bu[c] * w3s[h * KAN_NCOEF + c] * sc;
    xd[(size_t)h * D] = kan_leaky(v);
  }
}

// ---------------------------------------------------------------------------
extern "C" void kernel_launch(void* const* d_in, const int* in_sizes, int n_in,
                              void* d_out, int out_size, void* d_ws, size_t ws_size,
                              hipStream_t stream) {
  (void)in_sizes; (void)n_in; (void)out_size; (void)ws_size;

  const float* x   = (const float*)d_in[0];
  const float* w1b = (const float*)d_in[1];
  const float* w1s = (const float*)d_in[2];
  const float* w1c = (const float*)d_in[3];
  const float* w2b = (const float*)d_in[4];
  const float* w2s = (const float*)d_in[5];
  const float* w2c = (const float*)d_in[6];
  const float* w3b = (const float*)d_in[7];
  const float* w3s = (const float*)d_in[8];
  const float* w3c = (const float*)d_in[9];
  const float* w4b = (const float*)d_in[10];
  const float* w4s = (const float*)d_in[11];
  const float* w4c = (const float*)d_in[12];

  const int B = 64, H = 64, E = 1024, D = 1024;
  const int N = B * H;       // 4096 rows for the big GEMMs
  const int K = 9 * E;       // 9216 augmented inner dim (E == D)

  float* out    = (float*)d_out;
  float* x_dec  = out;                              // (B,H,E)
  float* x_emb  = out + (size_t)N * D;              // (B,H,D)
  float* x_de   = out + 2 * (size_t)N * D;          // (B,H,D)
  float* x_comb = out + 3 * (size_t)N * D;          // (B,1,D) flat = B*D

  // Workspace: A buffer (N x K f16, 75.5 MB) + W buffer (1024 x K f16, 18.9 MB),
  // reused sequentially for layers 1 and 4.
  _Float16* Abuf = (_Float16*)d_ws;
  _Float16* Wbuf = (_Float16*)((char*)d_ws + (size_t)N * K * sizeof(_Float16));

  dim3 blk(256);

  // ---- layer 1: E -> D, leaky-relu epilogue, writes x_emb ----
  kan_build_w_kernel<<<(D * E + 255) / 256, blk, 0, stream>>>(w1b, w1s, w1c, Wbuf, D, E);
  kan_build_a_kernel<<<(N * E + 255) / 256, blk, 0, stream>>>(x, Abuf, N, E);
  kan_wmma_gemm_kernel<<<dim3(D / 256, N / 128), blk, 0, stream>>>(
      Abuf, Wbuf, x_emb, N, D, K, /*leaky=*/1);

  // ---- layers 2+3 fused: writes x_comb and x_de ----
  kan_mid_kernel<<<(B * D + 255) / 256, blk, 0, stream>>>(
      x_emb, w2b, w2s, w2c, w3b, w3s, w3c, x_comb, x_de, B, H, D);

  // ---- layer 4: D -> E, no epilogue activation, writes x_dec ----
  kan_build_w_kernel<<<(E * D + 255) / 256, blk, 0, stream>>>(w4b, w4s, w4c, Wbuf, E, D);
  kan_build_a_kernel<<<(N * D + 255) / 256, blk, 0, stream>>>(x_de, Abuf, N, D);
  kan_wmma_gemm_kernel<<<dim3(E / 256, N / 128), blk, 0, stream>>>(
      Abuf, Wbuf, x_dec, N, E, K, /*leaky=*/0);
}
